// DCdetectorModel_73924977099336
// MI455X (gfx1250) — compile-verified, matrix operand order checked by reading, not run on hardware
//
#include <hip/hip_runtime.h>
#include <hip/hip_bf16.h>
#include <math.h>

// ---------------------------------------------------------------------------
// DCdetector forward for gfx1250 (MI455X), wave32 + WMMA f16 (16x16x32).
//
// Algebraic fusion (valid for H=1): with q=zWq+bq, k=zWk+bk,
//   S[t,s] = z_t (WqWk^T) z_s^T + z_t.(Wq bk) + z_s.(Wk bq) + bq.bk
// and the z_t.(..)+const terms are constant over the softmax axis s, so
//   softmax_s(scale*S) == softmax_s(scale*(z A z^T + 1.(z v)^T)),
//   A = WqWk^T (precomputed per layer), v = Wk bq.
// This halves the projection GEMM work and removes the K LDS buffer.
//
// Pipeline:
//   k_norm   : RevIN per-(b,m) standardization -> xt [7040,105] f32
//   k_wcvt   : Wq/Wk f32 -> f16 copies
//   k_prep   : MA[l] = Wk_l x Wq_l^T (row d' holds K=d of A^T) via WMMA
//   k_bias   : vb[l] = Wk_l . bq_l
//   k_attn<NMT>: per (config, sequence) block: conv embed + PE -> z(LDS,f16),
//              3 layers of { G=z*MA^T (WMMA, B-frag reused over NMT row
//              tiles), S=G*z^T/16 (WMMA), +col bias, row softmax, atomic
//              channel-mean accumulate }.  NMT=ceil(T/16): 3/2/1.
//   k_expand : repeat/tile expansion of channel means into d_out
// ---------------------------------------------------------------------------

typedef __attribute__((ext_vector_type(16))) _Float16 v16h;
typedef __attribute__((ext_vector_type(8)))  _Float16 v8h;
typedef __attribute__((ext_vector_type(8)))  float    v8f;

#define BN    128
#define LSEQ  105
#define MCH   55
#define DEMB  256
#define NLAY  3
#define NSEQ  (BN * MCH)      // 7040
#define ZSTR  264             // halves: LDS row stride for z/G (skews banks)
#define SSTR  49              // floats: LDS row stride for scores

// configs: 0:(p=3,size T=35) 1:(p=3,num T=3) 2:(p=5,size,21) 3:(p=5,num,5)
//          4:(p=7,size,15)   5:(p=7,num,7)
__constant__ int c_T[6]     = {35, 3, 21, 5, 15, 7};
__constant__ int c_aoff[6]  = {0, 470400, 473856, 643200, 652800, 739200};
__constant__ int c_cfg16[4] = {1, 3, 4, 5};   // cfgs handled by NMT=1 launch
#define ACC_TOTAL 758016     // floats: sum of 3*128*T*T over configs

// Per-instantiation LDS bytes: 512 + 2*TP*528 + TP*49*4 + TP*4
__host__ __device__ constexpr int smem_bytes(int nmt) {
  return 512 + 2 * (nmt * 16) * ZSTR * 2 + (nmt * 16) * SSTR * 4 + (nmt * 16) * 4;
}

// Load a 16-half WMMA fragment per lane: halves {0..7}=K[8h..8h+7],
// {8..15}=K[16+8h..] relative to p0 = base + row*stride + kb + 8*h.
__device__ __forceinline__ v16h frag16(const _Float16* p0) {
  v8h c0 = *(const v8h*)(p0);
  v8h c1 = *(const v8h*)(p0 + 16);
  v16h a;
#pragma unroll
  for (int i = 0; i < 8; ++i) { a[i] = c0[i]; a[i + 8] = c1[i]; }
  return a;
}

// --------------------------- RevIN normalize -------------------------------
__global__ void __launch_bounds__(128) k_norm(const float* __restrict__ x,
                                              float* __restrict__ xt) {
  __shared__ float s1[128], s2[128];
  int row = blockIdx.x;                       // b*55 + m
  int b = row / MCH, m = row - b * MCH;
  int t = threadIdx.x;
  float v = 0.f;
  if (t < LSEQ) v = x[((long long)b * LSEQ + t) * MCH + m];
  s1[t] = (t < LSEQ) ? v : 0.f;
  s2[t] = (t < LSEQ) ? v * v : 0.f;
  __syncthreads();
  for (int o = 64; o; o >>= 1) {
    if (t < o) { s1[t] += s1[t + o]; s2[t] += s2[t + o]; }
    __syncthreads();
  }
  float mean = s1[0] * (1.f / LSEQ);
  float var  = s2[0] * (1.f / LSEQ) - mean * mean;
  float rstd = rsqrtf(var + 1e-5f);
  if (t < LSEQ) xt[(long long)row * LSEQ + t] = (v - mean) * rstd;
}

// ---------------------- f32 -> f16 weight copies ---------------------------
__global__ void k_wcvt(const float* __restrict__ Wq, const float* __restrict__ Wk,
                       _Float16* __restrict__ wqh, _Float16* __restrict__ wkh) {
  int e = blockIdx.x * 256 + threadIdx.x;
  const int n1 = NLAY * DEMB * DEMB;
  if (e >= 2 * n1) return;
  if (e < n1) wqh[e] = (_Float16)Wq[e];
  else        wkh[e - n1] = (_Float16)Wk[e - n1];
}

// ---- MA[l][d'][d] = sum_e Wk[l][d'][e] * Wq[l][d][e]  (= A^T rows) --------
__global__ void __launch_bounds__(256) k_prep(const _Float16* __restrict__ wqh,
                                              const _Float16* __restrict__ wkh,
                                              _Float16* __restrict__ ma) {
  const int lane = threadIdx.x & 31;
  const int wv   = threadIdx.x >> 5;
  const int tile = blockIdx.x * 8 + wv;       // 0..767 (3 layers * 256 tiles)
  const int l  = tile >> 8;
  const int tl = tile & 255;
  const int mt = tl >> 4, nt = tl & 15;
  const int r = lane & 15, h = lane >> 4;
  const _Float16* A = wkh + l * DEMB * DEMB;  // rows d' over K=e
  const _Float16* B = wqh + l * DEMB * DEMB;  // rows d  over K=e (B = Wq^T cols)
  v8f acc = {};
  for (int kb = 0; kb < DEMB; kb += 32) {
    v16h af = frag16(A + (mt * 16 + r) * DEMB + kb + 8 * h);
    v16h bf = frag16(B + (nt * 16 + r) * DEMB + kb + 8 * h);
    acc = __builtin_amdgcn_wmma_f32_16x16x32_f16(
        false, af, false, bf, (short)0, acc, false, false);
  }
  _Float16* out = ma + l * DEMB * DEMB;
#pragma unroll
  for (int vv = 0; vv < 8; ++vv)
    out[(mt * 16 + vv + 8 * h) * DEMB + nt * 16 + r] = (_Float16)acc[vv];
}

// ---------------- vb[l][d] = sum_e Wk[l][d][e] * bq[l][e] ------------------
__global__ void __launch_bounds__(256) k_bias(const float* __restrict__ Wk,
                                              const float* __restrict__ bq,
                                              float* __restrict__ vb) {
  int l = blockIdx.x, d = threadIdx.x;
  const float* wrow = Wk + ((long long)l * DEMB + d) * DEMB;
  const float* br   = bq + l * DEMB;
  float a = 0.f;
  for (int e = 0; e < DEMB; ++e) a += wrow[e] * br[e];
  vb[l * DEMB + d] = a;
}

// --------------------- fused embed + attention -----------------------------
template <int NMT>
__global__ void __launch_bounds__(256) k_attn(
    const float* __restrict__ xt,
    const float* __restrict__ w0, const float* __restrict__ w1,
    const float* __restrict__ w2, const float* __restrict__ w3,
    const float* __restrict__ w4, const float* __restrict__ w5,
    const _Float16* __restrict__ ma, const float* __restrict__ vb,
    float* __restrict__ accum) {
  constexpr int TP = NMT * 16;
  extern __shared__ char smem[];
  float*    sx  = (float*)(smem);                                   // 105 f32
  _Float16* zs  = (_Float16*)(smem + 512);                          // TP*264 f16
  _Float16* gs  = (_Float16*)(smem + 512 + TP * ZSTR * 2);          // TP*264 f16
  float*    ss  = (float*)(smem + 512 + 2 * TP * ZSTR * 2);         // TP*49 f32
  float*    scv = (float*)(smem + 512 + 2 * TP * ZSTR * 2 + TP * SSTR * 4);

  int cfg;
  if (NMT == 3)      cfg = 0;
  else if (NMT == 2) cfg = 2;
  else               cfg = c_cfg16[blockIdx.y];
  const int bm  = blockIdx.x;
  const int b   = bm / MCH;
  const int T   = c_T[cfg];
  const int Cin = LSEQ / T;
  const float* wc = cfg == 0 ? w0 : cfg == 1 ? w1 : cfg == 2 ? w2
                  : cfg == 3 ? w3 : cfg == 4 ? w4 : w5;
  float* acc_base = accum + c_aoff[cfg];

  const int tid  = threadIdx.x;
  const int lane = tid & 31;
  const int wv   = tid >> 5;

  // stage normalized series row
  for (int i = tid; i < LSEQ; i += 256) sx[i] = xt[(long long)bm * LSEQ + i];
  __syncthreads();

  // circular conv (kernel 3, wrap pad) + sinusoidal PE -> z f16 in LDS,
  // rows T..TP-1 zero-padded so WMMA row tiles are safe.
  const float kln = -logf(10000.f) / (float)DEMB;
  for (int e = tid; e < TP * DEMB; e += 256) {
    int t = e >> 8, d = e & 255;
    float v = 0.f;
    if (t < T) {
      float a = 0.f;
      for (int k = 0; k < 3; ++k) {
        int tc = t + k - 1;
        tc = (tc < 0) ? tc + T : (tc >= T ? tc - T : tc);
        const float* xr = sx + tc * Cin;
        const float* wr = wc + (d * Cin) * 3 + k;     // w[d][c][k]
        for (int c = 0; c < Cin; ++c) a += xr[c] * wr[3 * c];
      }
      float div = __expf((float)(d & ~1) * kln);
      float arg = (float)t * div;
      v = a + ((d & 1) ? __cosf(arg) : __sinf(arg));
    }
    zs[t * ZSTR + d] = (_Float16)v;
  }
  __syncthreads();

  const int r = lane & 15;
  const int h = lane >> 4;

  for (int l = 0; l < NLAY; ++l) {
    // ---- G = z * A.  Column-tile outer: each loaded A^T B-fragment is
    //      reused across all NMT row tiles in registers (1 pass over MA). ----
    const _Float16* MAl = ma + l * DEMB * DEMB;
    for (int nt = wv; nt < 16; nt += 8) {
      v8f acc[NMT];
#pragma unroll
      for (int m = 0; m < NMT; ++m) acc[m] = (v8f){};
      for (int kb = 0; kb < DEMB; kb += 32) {
        v16h bf = frag16(MAl + (nt * 16 + r) * DEMB + kb + 8 * h);
#pragma unroll
        for (int m = 0; m < NMT; ++m) {
          v16h af = frag16(zs + (m * 16 + r) * ZSTR + kb + 8 * h);
          acc[m] = __builtin_amdgcn_wmma_f32_16x16x32_f16(
              false, af, false, bf, (short)0, acc[m], false, false);
        }
      }
#pragma unroll
      for (int m = 0; m < NMT; ++m)
#pragma unroll
        for (int vv = 0; vv < 8; ++vv)
          gs[(m * 16 + vv + 8 * h) * ZSTR + nt * 16 + r] = (_Float16)acc[m][vv];
    }
    __syncthreads();

    // ---- S = (1/16) * G * z^T  (B-frag rows = z rows, reused over NMT) ----
    for (int nt = wv; nt < NMT; nt += 8) {
      v8f acc[NMT];
#pragma unroll
      for (int m = 0; m < NMT; ++m) acc[m] = (v8f){};
      for (int kb = 0; kb < DEMB; kb += 32) {
        v16h bf = frag16(zs + (nt * 16 + r) * ZSTR + kb + 8 * h);
#pragma unroll
        for (int m = 0; m < NMT; ++m) {
          v16h af = frag16(gs + (m * 16 + r) * ZSTR + kb + 8 * h);
          acc[m] = __builtin_amdgcn_wmma_f32_16x16x32_f16(
              false, af, false, bf, (short)0, acc[m], false, false);
        }
      }
#pragma unroll
      for (int m = 0; m < NMT; ++m)
#pragma unroll
        for (int vv = 0; vv < 8; ++vv)
          ss[(m * 16 + vv + 8 * h) * SSTR + nt * 16 + r] = acc[m][vv] * 0.0625f;
    }
    __syncthreads();

    // ---- column bias scv[s] = (1/16) * z_s . vb_l  (exact w/ biases) ----
    const float* vbl = vb + l * DEMB;
    for (int s = wv; s < T; s += 8) {
      float a = 0.f;
      for (int d = lane; d < DEMB; d += 32)
        a += (float)zs[s * ZSTR + d] * vbl[d];
      for (int off = 16; off; off >>= 1) a += __shfl_xor(a, off);
      if (lane == 0) scv[s] = a * 0.0625f;
    }
    __syncthreads();

    // ---- row softmax (valid region only) + channel-mean accumulate ----
    for (int t = wv; t < T; t += 8) {
      float mx = -3.0e38f;
      for (int s = lane; s < T; s += 32)
        mx = fmaxf(mx, ss[t * SSTR + s] + scv[s]);
      for (int off = 16; off; off >>= 1) mx = fmaxf(mx, __shfl_xor(mx, off));
      float sm = 0.f;
      for (int s = lane; s < T; s += 32)
        sm += __expf(ss[t * SSTR + s] + scv[s] - mx);
      for (int off = 16; off; off >>= 1) sm += __shfl_xor(sm, off);
      float inv = 1.f / sm;
      float* ap = acc_base + (((long long)l * BN + b) * T + t) * T;
      for (int s = lane; s < T; s += 32)
        atomicAdd(ap + s, __expf(ss[t * SSTR + s] + scv[s] - mx) * inv);
    }
    __syncthreads();
  }
}

// ----------------------- repeat/tile expansion -----------------------------
__global__ void k_expand(const float* __restrict__ accum, float* __restrict__ out) {
  const long long half = 9LL * BN * LSEQ * LSEQ;     // 12,700,800
  long long idx = (long long)blockIdx.x * blockDim.x + threadIdx.x;
  if (idx >= 2 * half) return;
  int isPrior = idx >= half;
  long long e = isPrior ? idx - half : idx;
  int s = (int)(e % LSEQ); e /= LSEQ;
  int t = (int)(e % LSEQ); e /= LSEQ;
  int b = (int)(e % BN);   e /= BN;
  int g = (int)e;                                    // 0..8, patch-major
  int i = g / 3, l = g - 3 * i;
  int p = (i == 0) ? 3 : (i == 1) ? 5 : 7;
  int n = LSEQ / p;
  int cfg, T, at, as;
  if (!isPrior) { cfg = 2 * i;     T = n; at = t / p; as = s / p; }  // repeat
  else          { cfg = 2 * i + 1; T = p; at = t % p; as = s % p; }  // tile
  float vv = accum[c_aoff[cfg] + (((long long)l * BN + b) * T + at) * T + as];
  out[idx] = vv * (1.0f / (float)MCH);
}

// ---------------------------------------------------------------------------
extern "C" void kernel_launch(void* const* d_in, const int* in_sizes, int n_in,
                              void* d_out, int out_size, void* d_ws, size_t ws_size,
                              hipStream_t stream) {
  (void)in_sizes; (void)n_in; (void)out_size; (void)ws_size;
  const float* x   = (const float*)d_in[0];
  const float* cp0 = (const float*)d_in[1];
  const float* cn0 = (const float*)d_in[2];
  const float* cp1 = (const float*)d_in[3];
  const float* cn1 = (const float*)d_in[4];
  const float* cp2 = (const float*)d_in[5];
  const float* cn2 = (const float*)d_in[6];
  const float* Wq  = (const float*)d_in[7];
  const float* bq  = (const float*)d_in[8];
  const float* Wk  = (const float*)d_in[9];
  const float* bk  = (const float*)d_in[10];
  (void)bk;  // bk-dependent terms are constant along softmax axis -> cancel

  char* ws = (char*)d_ws;
  float*    xt    = (float*)(ws + 0);            // 7040*105*4   = 2,956,800 B
  _Float16* wqh   = (_Float16*)(ws + 2956800);   // 3*256*256*2  =   393,216 B
  _Float16* wkh   = (_Float16*)(ws + 3350016);   //              =   393,216 B
  _Float16* ma    = (_Float16*)(ws + 3743232);   //              =   393,216 B
  float*    vb    = (float*)(ws + 4136448);      // 3*256*4      =     3,072 B
  float*    accum = (float*)(ws + 4139520);      // 758,016 f32  = 3,032,064 B

  hipMemsetAsync(accum, 0, ACC_TOTAL * sizeof(float), stream);
  k_norm<<<NSEQ, 128, 0, stream>>>(x, xt);
  k_wcvt<<<(2 * NLAY * DEMB * DEMB + 255) / 256, 256, 0, stream>>>(Wq, Wk, wqh, wkh);
  k_prep<<<(NLAY * 256) / 8, 256, 0, stream>>>(wqh, wkh, ma);
  k_bias<<<NLAY, DEMB, 0, stream>>>(Wk, bq, vb);

  dim3 g1(NSEQ, 1);
  k_attn<3><<<g1, 256, smem_bytes(3), stream>>>(xt, cp0, cn0, cp1, cn1, cp2, cn2,
                                                ma, vb, accum);   // cfg 0 (T=35)
  k_attn<2><<<g1, 256, smem_bytes(2), stream>>>(xt, cp0, cn0, cp1, cn1, cp2, cn2,
                                                ma, vb, accum);   // cfg 2 (T=21)
  dim3 g4(NSEQ, 4);
  k_attn<1><<<g4, 256, smem_bytes(1), stream>>>(xt, cp0, cn0, cp1, cn1, cp2, cn2,
                                                ma, vb, accum);   // cfgs 1,3,4,5

  long long tot = 2LL * 9 * BN * LSEQ * LSEQ;
  k_expand<<<(unsigned)((tot + 255) / 256), 256, 0, stream>>>(accum, (float*)d_out);
}
